// RuleE_2800318676993
// MI455X (gfx1250) — matrix operand order repaired and checked by reading, not compiled, not run on hardware
//
#include <hip/hip_runtime.h>
#include <math.h>

#define BATCH   32
#define MAXPV   50
#define MPAD    64
#define RULENUM 256

typedef _Float16 v16h __attribute__((ext_vector_type(16)));
typedef float    v8f  __attribute__((ext_vector_type(8)));

struct __align__(8) h4_t { _Float16 v[4]; };

// ---------------------------------------------------------------- prep: ra_i[r] = l2norm(rule_emb[r]) @ wI[128:] + bI
__global__ void k_prep_ra(const float* __restrict__ remb,
                          const float* __restrict__ w1, const float* __restrict__ b1,
                          const float* __restrict__ w2, const float* __restrict__ b2,
                          const float* __restrict__ w3, const float* __restrict__ b3,
                          float* ra1, float* ra2, float* ra3)
{
    int r = blockIdx.x, tid = threadIdx.x;
    __shared__ float rn[128]; __shared__ float red[128];
    float x = remb[r*128 + tid];
    red[tid] = x*x; __syncthreads();
    for (int s = 64; s > 0; s >>= 1) { if (tid < s) red[tid] += red[tid+s]; __syncthreads(); }
    float n = fmaxf(sqrtf(red[0]), 1e-12f);
    rn[tid] = x / n; __syncthreads();
    float a1 = b1[tid], a2 = b2[tid], a3 = b3[tid];
    #pragma unroll 4
    for (int k = 0; k < 128; ++k) {
        float rk = rn[k];
        a1 += rk * w1[(128+k)*128 + tid];
        a2 += rk * w2[(128+k)*128 + tid];
        a3 += rk * w3[(128+k)*128 + tid];
    }
    ra1[r*128+tid] = a1; ra2[r*128+tid] = a2; ra3[r*128+tid] = a3;
}

// ------------------------------------------- prep: gathered pa_i[b][t] = l2norm(E[prop]) @ wI[:128], vemb = l2norm(E[val])
__global__ void k_prep_pa(const int* __restrict__ prop, const int* __restrict__ val,
                          const float* __restrict__ etab,
                          const float* __restrict__ w1, const float* __restrict__ w2,
                          const float* __restrict__ w3,
                          float* pa1, float* pa2, float* pa3, float* vemb)
{
    int t = blockIdx.x, b = blockIdx.y, tid = threadIdx.x;
    int row = (b*MPAD + t)*128 + tid;
    if (t >= MAXPV) { pa1[row]=0.f; pa2[row]=0.f; pa3[row]=0.f; vemb[row]=0.f; return; }
    __shared__ float pe[128]; __shared__ float red[128];
    int e = prop[b*MAXPV + t];
    float x = etab[e*128 + tid];
    red[tid] = x*x; __syncthreads();
    for (int s = 64; s > 0; s >>= 1) { if (tid < s) red[tid] += red[tid+s]; __syncthreads(); }
    float n = fmaxf(sqrtf(red[0]), 1e-12f);
    pe[tid] = x / n;
    __syncthreads();
    int ev = val[b*MAXPV + t];
    float y = etab[ev*128 + tid];
    red[tid] = y*y; __syncthreads();
    for (int s = 64; s > 0; s >>= 1) { if (tid < s) red[tid] += red[tid+s]; __syncthreads(); }
    float nv = fmaxf(sqrtf(red[0]), 1e-12f);
    vemb[row] = y / nv;
    float a1 = 0.f, a2 = 0.f, a3 = 0.f;
    #pragma unroll 4
    for (int k = 0; k < 128; ++k) {
        float pk = pe[k];
        a1 += pk * w1[k*128 + tid];
        a2 += pk * w2[k*128 + tid];
        a3 += pk * w3[k*128 + tid];
    }
    pa1[row] = a1; pa2[row] = a2; pa3[row] = a3;
}

// ---------------------------------------------------------------- prep: paq[q] = l2norm(E[q]) @ fc1_w[:128]   (q < 256)
__global__ void k_prep_paq(const float* __restrict__ etab, const float* __restrict__ w1,
                           float* paq)
{
    int q = blockIdx.x, tid = threadIdx.x;
    __shared__ float pe[128]; __shared__ float red[128];
    float x = etab[q*128 + tid];
    red[tid] = x*x; __syncthreads();
    for (int s = 64; s > 0; s >>= 1) { if (tid < s) red[tid] += red[tid+s]; __syncthreads(); }
    float n = fmaxf(sqrtf(red[0]), 1e-12f);
    pe[tid] = x / n; __syncthreads();
    float a = 0.f;
    #pragma unroll 4
    for (int k = 0; k < 128; ++k) a += pe[k] * w1[k*128 + tid];
    paq[q*128 + tid] = a;
}

// ---------------------------------------------------------------- prep: fc6t[n][k] = f16(fc6_w[k][n])
__global__ void k_prep_fc6t(const float* __restrict__ fc6w, _Float16* __restrict__ fc6t)
{
    int i = blockIdx.x * blockDim.x + threadIdx.x;   // 0..16383
    int n = i >> 7, k = i & 127;
    fc6t[i] = (_Float16)fc6w[k*128 + n];
}

// ---------------------------------------------------------------- pi[r][q] -> rule_mask[r], regpart[r]  (deterministic sums)
__global__ void k_pi(const float* __restrict__ paq, const float* __restrict__ ra1,
                     const float* __restrict__ fc4w, const float* __restrict__ fc4b,
                     float* rule_mask, float* regpart)
{
    int r = blockIdx.x, tid = threadIdx.x;
    int wave = tid >> 5, lane = tid & 31;
    __shared__ float ra[128], w4[128];
    __shared__ float wmask[8], wreg[8];
    if (tid < 128) { ra[tid] = ra1[r*128 + tid]; w4[tid] = fc4w[tid]; }
    __syncthreads();
    float b4 = fc4b[0];
    float macc = 0.f, racc = 0.f;
    for (int q = wave; q < 256; q += 8) {
        float acc = 0.f;
        #pragma unroll
        for (int kk = 0; kk < 4; ++kk) {
            int k = lane + kk*32;
            float h = fmaxf(paq[q*128 + k] + ra[k], 0.f);
            acc += h * w4[k];
        }
        #pragma unroll
        for (int off = 16; off > 0; off >>= 1) acc += __shfl_xor(acc, off, 32);
        float pi = 1.0f / (1.0f + expf(-(acc + b4)));
        macc += (pi > 0.5f) ? pi : 0.f;
        racc += (q == r) ? (-logf(pi + 1e-8f) * 256.0f) : (-logf(1.0f - pi + 1e-8f));
    }
    if (lane == 0) { wmask[wave] = macc; wreg[wave] = racc; }
    __syncthreads();
    if (tid == 0) {
        float m = 0.f, g = 0.f;
        for (int w = 0; w < 8; ++w) { m += wmask[w]; g += wreg[w]; }
        rule_mask[r] = m + 1.0f;
        regpart[r]   = g;
    }
}

// ---------------------------------------------------------------- fused main: per (b, 8 rules): WMMA GEMM + score
__global__ void __launch_bounds__(256) k_main(
    const float* __restrict__ pa1, const float* __restrict__ pa2, const float* __restrict__ pa3,
    const float* __restrict__ ra1, const float* __restrict__ ra2, const float* __restrict__ ra3,
    const _Float16* __restrict__ fc6t, const float* __restrict__ fc6b,
    const float* __restrict__ fc4w, const float* __restrict__ fc4b,
    const float* __restrict__ fc5w, const float* __restrict__ fc5b,
    const float* __restrict__ vemb,
    const int* __restrict__ samemask, const int* __restrict__ padmask,
    const float* __restrict__ rule_mask, float* __restrict__ score)
{
    const int b = blockIdx.y;
    const int rbase = blockIdx.x * 8;
    const int tid = threadIdx.x;
    const int wave = tid >> 5, lane = tid & 31;
    const int lh = lane >> 4;          // wave half (K-split per ISA layout)
    const int l15 = lane & 15;

    extern __shared__ char smem[];
    float* sRa1 = (float*)smem;          // 128
    float* sRa2 = sRa1 + 128;
    float* sRa3 = sRa2 + 128;
    float* sW4  = sRa3 + 128;
    float* sW5  = sW4 + 128;
    float* sNq  = sW5 + 128;             // [64][4] ||g||^2 quarter partials
    float* sDq  = sNq + 256;             // [64][4] v.g
    float* sAq  = sDq + 256;             // [64][4] s1 logit partials
    float* sPq  = sAq + 256;             // [64][4] p logit partials
    float* sG   = sPq + 256;             // [64][132] g tile (biased), padded stride
    _Float16* sA = (_Float16*)(sG + 64*132);   // [64][136] relu tile, f16

    if (tid < 128) { sW4[tid] = fc4w[tid]; sW5[tid] = fc5w[tid]; }

    // per-wave B fragments straight from global fc6^T (block-invariant, live in VGPRs)
    // lane = column n, halves = run of 16 K values (contiguous k order, same as A)
    const int ncol = wave*16 + l15;
    v16h bfrag[4];
    #pragma unroll
    for (int kt = 0; kt < 4; ++kt) {
        const _Float16* bp = fc6t + ncol*128 + kt*32 + lh*16;
        #pragma unroll
        for (int h = 0; h < 16; ++h) bfrag[kt][h] = bp[h];
    }
    const float bias = fc6b[ncol];
    const float b4 = fc4b[0], b5 = fc5b[0];
    __syncthreads();

    for (int ri = 0; ri < 8; ++ri) {
        const int r = rbase + ri;
        if (tid < 128) {
            sRa1[tid] = ra1[r*128 + tid];
            sRa2[tid] = ra2[r*128 + tid];
            sRa3[tid] = ra3[r*128 + tid];
        }
        __syncthreads();

        // A tile: f16(relu(pa3[b] + ra3[r])), 64x128 @ stride 136 (vectorized b128 load / b64 store)
        #pragma unroll
        for (int it = 0; it < 8; ++it) {
            int idx = tid + it*256;            // 0..2047 groups of 4
            int m = idx >> 5, k4 = (idx & 31) << 2;
            const float4 v = *(const float4*)(pa3 + (b*MPAD + m)*128 + k4);
            h4_t hh;
            hh.v[0] = (_Float16)fmaxf(v.x + sRa3[k4+0], 0.f);
            hh.v[1] = (_Float16)fmaxf(v.y + sRa3[k4+1], 0.f);
            hh.v[2] = (_Float16)fmaxf(v.z + sRa3[k4+2], 0.f);
            hh.v[3] = (_Float16)fmaxf(v.w + sRa3[k4+3], 0.f);
            *(h4_t*)(sA + m*136 + k4) = hh;
        }
        __syncthreads();

        // GEMM: wave owns 16 columns; 4 M-tiles x 4 K-steps of v_wmma_f32_16x16x32_f16
        #pragma unroll
        for (int mt = 0; mt < 4; ++mt) {
            v8f acc = {0.f,0.f,0.f,0.f,0.f,0.f,0.f,0.f};
            #pragma unroll
            for (int kt = 0; kt < 4; ++kt) {
                v16h af;
                const _Float16* arow = sA + (mt*16 + l15)*136 + kt*32 + lh*16;
                #pragma unroll
                for (int h = 0; h < 8; ++h) af[h]   = arow[h];
                #pragma unroll
                for (int h = 0; h < 8; ++h) af[8+h] = arow[8 + h];
                acc = __builtin_amdgcn_wmma_f32_16x16x32_f16(false, af, false, bfrag[kt],
                                                             (short)0, acc, false, false);
            }
            // D layout: VGPR j -> row m = mt*16 + j + 8*lh, col = ncol; spill biased g to LDS
            #pragma unroll
            for (int j = 0; j < 8; ++j) {
                int m = mt*16 + j + lh*8;
                sG[m*132 + ncol] = acc[j] + bias;
            }
        }
        __syncthreads();

        // fused epilogue: thread (m, q) owns one contiguous 32-wide quarter row.
        // accumulates ||g||^2, v.g, and the s1/p logit partials in one pass (no shuffles).
        {
            const int m = tid >> 2, q = tid & 3;
            const float* grow = sG   + m*132 + q*32;
            const float* vrow = vemb + (b*MPAD + m)*128 + q*32;
            const float* p1r  = pa1  + (b*MPAD + m)*128 + q*32;
            const float* p2r  = pa2  + (b*MPAD + m)*128 + q*32;
            float ns = 0.f, dsum = 0.f, a1 = 0.f, a2 = 0.f;
            #pragma unroll
            for (int i4 = 0; i4 < 8; ++i4) {
                const int k = q*32 + i4*4;
                const float4 g4 = *(const float4*)(grow + i4*4);
                const float4 v4 = *(const float4*)(vrow + i4*4);
                const float4 x1 = *(const float4*)(p1r  + i4*4);
                const float4 x2 = *(const float4*)(p2r  + i4*4);
                ns   += g4.x*g4.x + g4.y*g4.y + g4.z*g4.z + g4.w*g4.w;
                dsum += g4.x*v4.x + g4.y*v4.y + g4.z*v4.z + g4.w*v4.w;
                a1 += fmaxf(x1.x + sRa1[k+0], 0.f) * sW4[k+0];
                a1 += fmaxf(x1.y + sRa1[k+1], 0.f) * sW4[k+1];
                a1 += fmaxf(x1.z + sRa1[k+2], 0.f) * sW4[k+2];
                a1 += fmaxf(x1.w + sRa1[k+3], 0.f) * sW4[k+3];
                a2 += fmaxf(x2.x + sRa2[k+0], 0.f) * sW5[k+0];
                a2 += fmaxf(x2.y + sRa2[k+1], 0.f) * sW5[k+1];
                a2 += fmaxf(x2.z + sRa2[k+2], 0.f) * sW5[k+2];
                a2 += fmaxf(x2.w + sRa2[k+3], 0.f) * sW5[k+3];
            }
            sNq[m*4+q] = ns; sDq[m*4+q] = dsum; sAq[m*4+q] = a1; sPq[m*4+q] = a2;
        }
        __syncthreads();

        // score[b][r] = sum_t(masked score) / rule_mask[r]
        if (wave == 0) {
            float accs = 0.f;
            for (int m = lane; m < MAXPV; m += 32) {
                float ns = sNq[m*4+0] + sNq[m*4+1] + sNq[m*4+2] + sNq[m*4+3];
                float ds = sDq[m*4+0] + sDq[m*4+1] + sDq[m*4+2] + sDq[m*4+3];
                float a1 = sAq[m*4+0] + sAq[m*4+1] + sAq[m*4+2] + sAq[m*4+3];
                float a2 = sPq[m*4+0] + sPq[m*4+1] + sPq[m*4+2] + sPq[m*4+3];
                float s1 = 1.0f / (1.0f + expf(-(a1 + b4)));
                float p  = 1.0f / (1.0f + expf(-(a2 + b5)));
                float cosv = ds / fmaxf(sqrtf(ns), 1e-12f);
                float s2 = 0.5f * cosv + 0.5f;
                float sc = p + (1.0f - p) * s2;
                sc = (samemask[b*MAXPV + m] == 1) ? (s1 * sc) : (1.0f - s1);
                sc *= (float)padmask[b*MAXPV + m];
                sc *= (s1 > 0.5f) ? 1.0f : 0.f;
                accs += sc;
            }
            #pragma unroll
            for (int off = 16; off > 0; off >>= 1) accs += __shfl_xor(accs, off, 32);
            if (lane == 0) score[b*RULENUM + r] = accs / rule_mask[r];
        }
        __syncthreads();
    }
}

// ---------------------------------------------------------------- finalize: max over rules + reg mean
__global__ void k_finalize(const float* __restrict__ score, const float* __restrict__ regpart,
                           float* __restrict__ out)
{
    int tid = threadIdx.x;
    if (tid < 32) {
        float mx = -3.4e38f;
        for (int r = 0; r < 256; ++r) mx = fmaxf(mx, score[tid*256 + r]);
        out[tid] = mx;
    } else if (tid == 32) {
        float s = 0.f;
        for (int r = 0; r < 256; ++r) s += regpart[r];
        out[32] = s * (1.0f / 65536.0f);
    }
}

// ---------------------------------------------------------------- host
extern "C" void kernel_launch(void* const* d_in, const int* in_sizes, int n_in,
                              void* d_out, int out_size, void* d_ws, size_t ws_size,
                              hipStream_t stream)
{
    (void)in_sizes; (void)n_in; (void)out_size; (void)ws_size;
    const int*   prop = (const int*)d_in[0];
    const int*   val  = (const int*)d_in[1];
    const int*   same = (const int*)d_in[2];
    const int*   pad  = (const int*)d_in[3];
    const float* remb = (const float*)d_in[4];
    const float* etab = (const float*)d_in[5];
    const float* fc1w = (const float*)d_in[6];
    const float* fc1b = (const float*)d_in[7];
    const float* fc2w = (const float*)d_in[8];
    const float* fc2b = (const float*)d_in[9];
    const float* fc3w = (const float*)d_in[10];
    const float* fc3b = (const float*)d_in[11];
    const float* fc4w = (const float*)d_in[12];
    const float* fc4b = (const float*)d_in[13];
    const float* fc5w = (const float*)d_in[14];
    const float* fc5b = (const float*)d_in[15];
    const float* fc6w = (const float*)d_in[16];
    const float* fc6b = (const float*)d_in[17];

    char* ws = (char*)d_ws;
    float*    ra1      = (float*)(ws + 0);
    float*    ra2      = (float*)(ws + 131072);
    float*    ra3      = (float*)(ws + 262144);
    float*    pa1      = (float*)(ws + 393216);
    float*    pa2      = (float*)(ws + 1441792);
    float*    pa3      = (float*)(ws + 2490368);
    float*    vemb     = (float*)(ws + 3538944);
    float*    paq      = (float*)(ws + 4587520);
    _Float16* fc6t     = (_Float16*)(ws + 4718592);
    float*    scorebuf = (float*)(ws + 4751360);
    float*    rulemask = (float*)(ws + 4784128);
    float*    regpart  = (float*)(ws + 4785152);

    k_prep_ra<<<RULENUM, 128, 0, stream>>>(remb, fc1w, fc1b, fc2w, fc2b, fc3w, fc3b,
                                           ra1, ra2, ra3);
    k_prep_pa<<<dim3(MPAD, BATCH), 128, 0, stream>>>(prop, val, etab, fc1w, fc2w, fc3w,
                                                     pa1, pa2, pa3, vemb);
    k_prep_paq<<<RULENUM, 128, 0, stream>>>(etab, fc1w, paq);
    k_prep_fc6t<<<64, 256, 0, stream>>>(fc6w, fc6t);
    k_pi<<<RULENUM, 256, 0, stream>>>(paq, ra1, fc4w, fc4b, rulemask, regpart);

    size_t smem = (size_t)(128*5 + 256*4 + 64*132) * sizeof(float)
                + (size_t)(64*136) * sizeof(_Float16);   // 57856 B
    k_main<<<dim3(32, BATCH), 256, smem, stream>>>(pa1, pa2, pa3, ra1, ra2, ra3,
                                                   fc6t, fc6b, fc4w, fc4b, fc5w, fc5b,
                                                   vemb, same, pad, rulemask, scorebuf);
    k_finalize<<<1, 64, 0, stream>>>(scorebuf, regpart, (float*)d_out);
}
